// MultiHeadAttention_77111842832996
// MI455X (gfx1250) — compile-verified, matrix operand order checked by reading, not run on hardware
//
#include <hip/hip_runtime.h>
#include <math.h>

// ---------------- problem constants (match reference) ----------------
constexpr int kB = 8;
constexpr int kS = 1024;
constexpr int kE = 768;
constexpr int kH = 12;
constexpr int kD = 64;   // kE / kH

typedef __bf16 bf16_t;
typedef bf16_t v16bf __attribute__((ext_vector_type(16)));
typedef bf16_t v8bf  __attribute__((ext_vector_type(8)));
typedef float  v8f   __attribute__((ext_vector_type(8)));

#define WMMA_BF16(a, b, c) \
    __builtin_amdgcn_wmma_f32_16x16x32_bf16(false, (a), false, (b), (short)0, (c), false, false)

// Load a 16-lane-wide A/B fragment for v_wmma_*_16x16x32_bf16.
// Row-major source, `row` is the M (A) or N (B) index for this lane,
// ld = leading dimension in elements, kbase = starting K (multiple of 32).
// ISA layout (16-bit, 16x32): lanes 0-15 hold K = kbase+{0..7, 16..23},
// lanes 16-31 hold K = kbase+{8..15, 24..31}  -> two b128 loads per lane.
__device__ inline v16bf load_frag(const bf16_t* __restrict__ base, int row, int ld,
                                  int kbase, int lane) {
    const int g = (lane >> 4) & 1;
    const bf16_t* p = base + (size_t)row * ld + kbase + g * 8;
    v8bf lo = *(const v8bf*)(p);        // K = kbase + g*8 .. +7
    v8bf hi = *(const v8bf*)(p + 16);   // K = kbase + 16 + g*8 .. +7
    v16bf r;
#pragma unroll
    for (int i = 0; i < 8; ++i) { r[i] = lo[i]; r[8 + i] = hi[i]; }
    return r;
}

// ---------------- stage 1: conversions ----------------
__global__ void f32_to_bf16_kernel(const float* __restrict__ src,
                                   bf16_t* __restrict__ dst, int n) {
    int i = blockIdx.x * blockDim.x + threadIdx.x;
    if (i < n) dst[i] = (bf16_t)src[i];
}

// src [Hn, R, C] fp32 -> dst [Hn, C, R] bf16 (weight transpose so B-fragments
// get contiguous K-runs).
__global__ void cvt_transpose_kernel(const float* __restrict__ src,
                                     bf16_t* __restrict__ dst,
                                     int Hn, int R, int C) {
    int i = blockIdx.x * blockDim.x + threadIdx.x;
    int tot = Hn * R * C;
    if (i >= tot) return;
    int hh = i / (R * C);
    int r  = (i / C) % R;
    int c  = i % C;
    dst[(size_t)hh * R * C + (size_t)c * R + r] = (bf16_t)src[i];
}

// ---------------- stage 2: fused QKV projection ----------------
// One wave computes a 32(M) x 64(N=D) tile of X[b] (SxE) @ W[h] + bias.
// All 6 fragments of a k-step are loaded into distinct registers up front
// (one load clause), then 8 WMMAs drain them with progressive loadcnt waits.
// grid = (S/32, 3, B*H); blockIdx.y selects Q/K/V. V is stored transposed
// [BH, D, S] with packed 16-byte stores.
__global__ void __launch_bounds__(32)
qkv_proj_kernel(const bf16_t* __restrict__ X,    // [B*S, E] bf16
                const bf16_t* __restrict__ Wqt,  // [H, D, E] bf16 (pre-transposed)
                const bf16_t* __restrict__ Wkt,
                const bf16_t* __restrict__ Wvt,
                const float*  __restrict__ bq,   // [H, D] fp32
                const float*  __restrict__ bk,
                const float*  __restrict__ bv,
                bf16_t* __restrict__ Qb,         // [BH, S, D]
                bf16_t* __restrict__ Kb,         // [BH, S, D]
                bf16_t* __restrict__ Vtb) {      // [BH, D, S]
    const int tileS = blockIdx.x;
    const int which = blockIdx.y;   // 0=Q, 1=K, 2=V
    const int bh    = blockIdx.z;
    const int b     = bh / kH;
    const int h     = bh % kH;
    const int lane  = threadIdx.x;
    const int g     = lane >> 4;
    const int m     = lane & 15;

    const bf16_t* Wt   = (which == 0) ? Wqt : (which == 1) ? Wkt : Wvt;
    const float*  bias = (which == 0) ? bq  : (which == 1) ? bk  : bv;

    const bf16_t* Xb = X  + (size_t)b * kS * kE;
    const bf16_t* Wh = Wt + (size_t)h * kD * kE;
    const int rowBase = tileS * 32;

    v8f acc[2][4];
#pragma unroll
    for (int mt = 0; mt < 2; ++mt)
#pragma unroll
        for (int n = 0; n < 4; ++n)
#pragma unroll
            for (int i = 0; i < 8; ++i) acc[mt][n][i] = 0.f;

    for (int k = 0; k < kE; k += 32) {
        // hoist all fragment loads: distinct registers -> one clause, no
        // false WAR dependence between loads and WMMAs
        v16bf a0 = load_frag(Xb, rowBase + m,      kE, k, lane);
        v16bf a1 = load_frag(Xb, rowBase + 16 + m, kE, k, lane);
        v16bf w0 = load_frag(Wh, 0 * 16 + m, kE, k, lane);
        v16bf w1 = load_frag(Wh, 1 * 16 + m, kE, k, lane);
        v16bf w2 = load_frag(Wh, 2 * 16 + m, kE, k, lane);
        v16bf w3 = load_frag(Wh, 3 * 16 + m, kE, k, lane);
        acc[0][0] = WMMA_BF16(a0, w0, acc[0][0]);
        acc[1][0] = WMMA_BF16(a1, w0, acc[1][0]);
        acc[0][1] = WMMA_BF16(a0, w1, acc[0][1]);
        acc[1][1] = WMMA_BF16(a1, w1, acc[1][1]);
        acc[0][2] = WMMA_BF16(a0, w2, acc[0][2]);
        acc[1][2] = WMMA_BF16(a1, w2, acc[1][2]);
        acc[0][3] = WMMA_BF16(a0, w3, acc[0][3]);
        acc[1][3] = WMMA_BF16(a1, w3, acc[1][3]);
    }

    float bvv[4];
#pragma unroll
    for (int n = 0; n < 4; ++n) bvv[n] = bias[h * kD + n * 16 + m];

    if (which != 2) {
        bf16_t* o = ((which == 0) ? Qb : Kb) + (size_t)bh * kS * kD;
#pragma unroll
        for (int mt = 0; mt < 2; ++mt)
#pragma unroll
            for (int n = 0; n < 4; ++n)
#pragma unroll
                for (int i = 0; i < 8; ++i) {
                    int row = rowBase + mt * 16 + i + 8 * g;
                    o[(size_t)row * kD + n * 16 + m] = (bf16_t)(acc[mt][n][i] + bvv[n]);
                }
    } else {  // V^T: [BH, D, S]; rows i+8g are contiguous -> packed b128 store
        bf16_t* o = Vtb + (size_t)bh * kD * kS;
#pragma unroll
        for (int mt = 0; mt < 2; ++mt)
#pragma unroll
            for (int n = 0; n < 4; ++n) {
                v8bf pk;
#pragma unroll
                for (int i = 0; i < 8; ++i) pk[i] = (bf16_t)(acc[mt][n][i] + bvv[n]);
                bf16_t* p = o + (size_t)(n * 16 + m) * kS + rowBase + mt * 16 + 8 * g;
                *(v8bf*)p = pk;
            }
    }
}

// ---------------- stage 3: flash attention ----------------
// One wave handles 32 query rows of one (b,h); streams keys 32 at a time with
// online softmax. Each K/V fragment feeds two WMMAs. K/V fragments for each
// phase are hoisted into distinct registers so loads clause together. Scores
// never touch HBM. Next K/V tiles prefetched via global_prefetch_b8.
__global__ void __launch_bounds__(32)
flash_attn_kernel(const bf16_t* __restrict__ Q,   // [BH, S, D]
                  const bf16_t* __restrict__ K,   // [BH, S, D]
                  const bf16_t* __restrict__ Vt,  // [BH, D, S]
                  bf16_t* __restrict__ attn) {    // [B, S, E] (heads concat)
    __shared__ __align__(16) bf16_t pbuf[32 * 32];  // P tile bounce buffer

    const int qt   = blockIdx.x;   // 32-row query tile
    const int bh   = blockIdx.y;
    const int b    = bh / kH;
    const int h    = bh % kH;
    const int lane = threadIdx.x;
    const int g    = lane >> 4;
    const int m    = lane & 15;

    const bf16_t* Qp = Q  + (size_t)bh * kS * kD;
    const bf16_t* Kp = K  + (size_t)bh * kS * kD;
    const bf16_t* Vp = Vt + (size_t)bh * kD * kS;

    // Q A-fragments (2 row sub-tiles x 2 k-steps over D=64), loaded once.
    v16bf qf[2][2];
#pragma unroll
    for (int mt = 0; mt < 2; ++mt)
#pragma unroll
        for (int kb = 0; kb < 2; ++kb)
            qf[mt][kb] = load_frag(Qp, qt * 32 + mt * 16 + m, kD, kb * 32, lane);

    float mi[2][8], li[2][8];
    v8f accd[2][4];
#pragma unroll
    for (int mt = 0; mt < 2; ++mt)
#pragma unroll
        for (int i = 0; i < 8; ++i) { mi[mt][i] = -INFINITY; li[mt][i] = 0.f; }
#pragma unroll
    for (int mt = 0; mt < 2; ++mt)
#pragma unroll
        for (int j = 0; j < 4; ++j)
#pragma unroll
            for (int i = 0; i < 8; ++i) accd[mt][j][i] = 0.f;

    const float scale = 0.125f;  // 1/sqrt(64)

    for (int t0 = 0; t0 < kS; t0 += 32) {
        // prefetch next key/value tiles (global_prefetch_b8)
        if (t0 + 32 < kS) {
            __builtin_prefetch(Kp + (size_t)(t0 + 32 + lane) * kD, 0, 1);
            __builtin_prefetch(Vp + (size_t)lane * kS + t0 + 32, 0, 1);
            __builtin_prefetch(Vp + (size_t)(lane + 32) * kS + t0 + 32, 0, 1);
        }

        // --- scores: hoist 4 K-fragments, then 8 WMMAs ---
        v16bf kf00 = load_frag(Kp, t0 + m,      kD, 0,  lane);
        v16bf kf01 = load_frag(Kp, t0 + m,      kD, 32, lane);
        v16bf kf10 = load_frag(Kp, t0 + 16 + m, kD, 0,  lane);
        v16bf kf11 = load_frag(Kp, t0 + 16 + m, kD, 32, lane);
        v8f s[2][2];
#pragma unroll
        for (int mt = 0; mt < 2; ++mt)
#pragma unroll
            for (int ct = 0; ct < 2; ++ct)
#pragma unroll
                for (int i = 0; i < 8; ++i) s[mt][ct][i] = 0.f;
        s[0][0] = WMMA_BF16(qf[0][0], kf00, s[0][0]);
        s[1][0] = WMMA_BF16(qf[1][0], kf00, s[1][0]);
        s[0][0] = WMMA_BF16(qf[0][1], kf01, s[0][0]);
        s[1][0] = WMMA_BF16(qf[1][1], kf01, s[1][0]);
        s[0][1] = WMMA_BF16(qf[0][0], kf10, s[0][1]);
        s[1][1] = WMMA_BF16(qf[1][0], kf10, s[1][1]);
        s[0][1] = WMMA_BF16(qf[0][1], kf11, s[0][1]);
        s[1][1] = WMMA_BF16(qf[1][1], kf11, s[1][1]);

        // --- online softmax over 32 new columns (row = i + 8*g, replicated
        //     across the 16 lanes of each half-wave) ---
        v8f p[2][2];
#pragma unroll
        for (int mt = 0; mt < 2; ++mt) {
#pragma unroll
            for (int i = 0; i < 8; ++i) {
                float a0 = s[mt][0][i] * scale;
                float a1 = s[mt][1][i] * scale;
                float v = fmaxf(a0, a1);
#pragma unroll
                for (int off = 1; off < 16; off <<= 1) v = fmaxf(v, __shfl_xor(v, off, 16));
                float mn = fmaxf(mi[mt][i], v);
                float alpha = __expf(mi[mt][i] - mn);
                mi[mt][i] = mn;
                float e0 = __expf(a0 - mn);
                float e1 = __expf(a1 - mn);
                p[mt][0][i] = e0;
                p[mt][1][i] = e1;
                float rs = e0 + e1;
#pragma unroll
                for (int off = 1; off < 16; off <<= 1) rs += __shfl_xor(rs, off, 16);
                li[mt][i] = li[mt][i] * alpha + rs;
#pragma unroll
                for (int j = 0; j < 4; ++j) accd[mt][j][i] *= alpha;
            }
        }

        // --- bounce P (32x32) through LDS to re-layout C-frag -> A-frag ---
#pragma unroll
        for (int mt = 0; mt < 2; ++mt)
#pragma unroll
            for (int i = 0; i < 8; ++i) {
                int row = mt * 16 + i + 8 * g;
                pbuf[row * 32 + m]      = (bf16_t)p[mt][0][i];
                pbuf[row * 32 + 16 + m] = (bf16_t)p[mt][1][i];
            }
        asm volatile("s_wait_dscnt 0" ::: "memory");
        v16bf pf0 = load_frag(pbuf, m,      32, 0, lane);
        v16bf pf1 = load_frag(pbuf, 16 + m, 32, 0, lane);

        // --- accd += P (32x32) @ V (32x64); hoist 4 V-fragments ---
        v16bf vf0 = load_frag(Vp, 0 * 16 + m, kS, t0, lane);
        v16bf vf1 = load_frag(Vp, 1 * 16 + m, kS, t0, lane);
        v16bf vf2 = load_frag(Vp, 2 * 16 + m, kS, t0, lane);
        v16bf vf3 = load_frag(Vp, 3 * 16 + m, kS, t0, lane);
        accd[0][0] = WMMA_BF16(pf0, vf0, accd[0][0]);
        accd[1][0] = WMMA_BF16(pf1, vf0, accd[1][0]);
        accd[0][1] = WMMA_BF16(pf0, vf1, accd[0][1]);
        accd[1][1] = WMMA_BF16(pf1, vf1, accd[1][1]);
        accd[0][2] = WMMA_BF16(pf0, vf2, accd[0][2]);
        accd[1][2] = WMMA_BF16(pf1, vf2, accd[1][2]);
        accd[0][3] = WMMA_BF16(pf0, vf3, accd[0][3]);
        accd[1][3] = WMMA_BF16(pf1, vf3, accd[1][3]);
    }

    // --- normalize and store (concat-head layout [B,S,E]) ---
    bf16_t* op = attn + ((size_t)(b * kS + qt * 32)) * kE + h * kD;
#pragma unroll
    for (int mt = 0; mt < 2; ++mt)
#pragma unroll
        for (int i = 0; i < 8; ++i) {
            float inv = 1.f / li[mt][i];
            int row = mt * 16 + i + 8 * g;
#pragma unroll
            for (int j = 0; j < 4; ++j)
                op[(size_t)row * kE + j * 16 + m] = (bf16_t)(accd[mt][j][i] * inv);
        }
}

// ---------------- stage 4: output projection ----------------
// out[B*S, E] = attn[B*S, E] @ Wo[E, E] + bo  (fp32 result)
// 32(M) x 64(N) per wave: 8 WMMAs per k-step, fragments hoisted into distinct
// registers for clause'd loads.
__global__ void __launch_bounds__(32)
out_proj_kernel(const bf16_t* __restrict__ A,    // [B*S, E]
                const bf16_t* __restrict__ Wot,  // [E, E] (row n = Wo[:, n])
                const float*  __restrict__ bo,   // [E]
                float* __restrict__ out) {       // [B*S, E]
    const int tr   = blockIdx.x;  // 32-row tile over B*S
    const int tc   = blockIdx.y;  // 64-col tile over E
    const int lane = threadIdx.x;
    const int g    = lane >> 4;
    const int m    = lane & 15;
    const int rowBase = tr * 32;

    v8f acc[2][4];
#pragma unroll
    for (int mt = 0; mt < 2; ++mt)
#pragma unroll
        for (int n = 0; n < 4; ++n)
#pragma unroll
            for (int i = 0; i < 8; ++i) acc[mt][n][i] = 0.f;

    for (int k = 0; k < kE; k += 32) {
        v16bf a0 = load_frag(A, rowBase + m,      kE, k, lane);
        v16bf a1 = load_frag(A, rowBase + 16 + m, kE, k, lane);
        v16bf w0 = load_frag(Wot, tc * 64 + 0 * 16 + m, kE, k, lane);
        v16bf w1 = load_frag(Wot, tc * 64 + 1 * 16 + m, kE, k, lane);
        v16bf w2 = load_frag(Wot, tc * 64 + 2 * 16 + m, kE, k, lane);
        v16bf w3 = load_frag(Wot, tc * 64 + 3 * 16 + m, kE, k, lane);
        acc[0][0] = WMMA_BF16(a0, w0, acc[0][0]);
        acc[1][0] = WMMA_BF16(a1, w0, acc[1][0]);
        acc[0][1] = WMMA_BF16(a0, w1, acc[0][1]);
        acc[1][1] = WMMA_BF16(a1, w1, acc[1][1]);
        acc[0][2] = WMMA_BF16(a0, w2, acc[0][2]);
        acc[1][2] = WMMA_BF16(a1, w2, acc[1][2]);
        acc[0][3] = WMMA_BF16(a0, w3, acc[0][3]);
        acc[1][3] = WMMA_BF16(a1, w3, acc[1][3]);
    }

#pragma unroll
    for (int n = 0; n < 4; ++n) {
        float bvv = bo[tc * 64 + n * 16 + m];
#pragma unroll
        for (int mt = 0; mt < 2; ++mt)
#pragma unroll
            for (int i = 0; i < 8; ++i) {
                int row = rowBase + mt * 16 + i + 8 * g;
                out[(size_t)row * kE + tc * 64 + n * 16 + m] = acc[mt][n][i] + bvv;
            }
    }
}

// ---------------- host-side launch ----------------
extern "C" void kernel_launch(void* const* d_in, const int* in_sizes, int n_in,
                              void* d_out, int out_size, void* d_ws, size_t ws_size,
                              hipStream_t stream) {
    (void)in_sizes; (void)n_in; (void)out_size; (void)ws_size;

    const float* hidden = (const float*)d_in[0];
    const float* Wq = (const float*)d_in[1];
    const float* bq = (const float*)d_in[2];
    const float* Wk = (const float*)d_in[3];
    const float* bk = (const float*)d_in[4];
    const float* Wv = (const float*)d_in[5];
    const float* bv = (const float*)d_in[6];
    const float* Wo = (const float*)d_in[7];
    const float* bo = (const float*)d_in[8];
    float* out = (float*)d_out;

    // workspace carve-up (bf16 elements)
    bf16_t* w = (bf16_t*)d_ws;
    size_t off = 0;
    bf16_t* Xbf = w + off; off += (size_t)kB * kS * kE;
    bf16_t* Wqt = w + off; off += (size_t)kH * kD * kE;
    bf16_t* Wkt = w + off; off += (size_t)kH * kD * kE;
    bf16_t* Wvt = w + off; off += (size_t)kH * kD * kE;
    bf16_t* Wot = w + off; off += (size_t)kE * kE;
    bf16_t* Qb  = w + off; off += (size_t)kB * kH * kS * kD;
    bf16_t* Kb  = w + off; off += (size_t)kB * kH * kS * kD;
    bf16_t* Vtb = w + off; off += (size_t)kB * kH * kS * kD;
    bf16_t* Att = w + off; off += (size_t)kB * kS * kE;

    // stage 1: conversions
    {
        int n = kB * kS * kE;
        f32_to_bf16_kernel<<<(n + 255) / 256, 256, 0, stream>>>(hidden, Xbf, n);
        int nw = kH * kE * kD;
        cvt_transpose_kernel<<<(nw + 255) / 256, 256, 0, stream>>>(Wq, Wqt, kH, kE, kD);
        cvt_transpose_kernel<<<(nw + 255) / 256, 256, 0, stream>>>(Wk, Wkt, kH, kE, kD);
        cvt_transpose_kernel<<<(nw + 255) / 256, 256, 0, stream>>>(Wv, Wvt, kH, kE, kD);
        int no = kE * kE;
        cvt_transpose_kernel<<<(no + 255) / 256, 256, 0, stream>>>(Wo, Wot, 1, kE, kE);
    }

    // stage 2: fused QKV projections (V stored transposed for PV fragments)
    {
        dim3 grid(kS / 32, 3, kB * kH);
        qkv_proj_kernel<<<grid, 32, 0, stream>>>(Xbf, Wqt, Wkt, Wvt,
                                                 bq, bk, bv, Qb, Kb, Vtb);
    }

    // stage 3: flash attention
    {
        dim3 grid(kS / 32, kB * kH);
        flash_attn_kernel<<<grid, 32, 0, stream>>>(Qb, Kb, Vtb, Att);
    }

    // stage 4: output projection
    {
        dim3 grid((kB * kS) / 32, kE / 64);
        out_proj_kernel<<<grid, 32, 0, stream>>>(Att, Wot, bo, out);
    }
}
